// InverseFoldingModel_36593121362115
// MI455X (gfx1250) — compile-verified, matrix operand order checked by reading, not run on hardware
//
#include <hip/hip_runtime.h>

// ---------------------------------------------------------------------------
// CDNA5 (gfx1250) bf16 WMMA implementation of the inverse-folding GNN.
// All GEMMs use v_wmma_f32_16x16x32_bf16; weights pre-packed into B-fragment
// layout; gathers feed A-fragments directly; LN + residual fused in epilogue.
// f32->bf16 conversions use native casts so the backend can emit
// v_cvt_pk_bf16_f32 instead of manual bit-twiddling.
// ---------------------------------------------------------------------------

typedef __attribute__((ext_vector_type(16))) __bf16 v16bf;
typedef __attribute__((ext_vector_type(8)))  float  v8f;
typedef __attribute__((ext_vector_type(4)))  float  vf4;
typedef __attribute__((ext_vector_type(4)))  __bf16 vbf4;

union FragU {
  v16bf v;
  uint4 u[2];
  unsigned short s[16];
};

__device__ __forceinline__ unsigned short f2bf(float f) {
  union { __bf16 h; unsigned short s; } u;
  u.h = (__bf16)f;   // native f32->bf16 (RNE) on gfx1250
  return u.s;
}
__device__ __forceinline__ float bf2f(unsigned short s) {
  return __uint_as_float(((unsigned int)s) << 16);
}
__device__ __forceinline__ v8f zero8() {
  v8f z = {0.f, 0.f, 0.f, 0.f, 0.f, 0.f, 0.f, 0.f};
  return z;
}
__device__ __forceinline__ v8f wmma_bf16(v16bf a, v16bf b, v8f c) {
  return __builtin_amdgcn_wmma_f32_16x16x32_bf16(false, a, false, b, (short)0, c,
                                                 false, false);
}
// A fragment: 16x32 bf16 tile row mrow; lane needs runs [k0,k0+8) and [k0+16,k0+24)
__device__ __forceinline__ v16bf load_afrag(const unsigned short* seg, int k0) {
  FragU A;
  A.u[0] = *(const uint4*)(seg + k0);
  A.u[1] = *(const uint4*)(seg + k0 + 16);
  return A.v;
}
// B fragment: pre-packed, 32 contiguous bytes per lane
__device__ __forceinline__ v16bf load_bfrag(const unsigned short* pw, int frag, int lane) {
  const uint4* p = (const uint4*)(pw + (((size_t)frag * 32) + lane) * 16);
  FragU B;
  B.u[0] = p[0];
  B.u[1] = p[1];
  return B.v;
}

// ---------------------------------------------------------------------------
// Weight packing: W (Kdim x 128 row-major f32) -> bf16 WMMA B fragments.
// Fragment f = kt*8 + t covers K rows [kt*32, kt*32+32), cols [t*16, t*16+16).
// ---------------------------------------------------------------------------
__global__ void pack_w_kernel(const float* __restrict__ W, int Kdim,
                              unsigned short* __restrict__ dst, int nfrag) {
  int tid = blockIdx.x * blockDim.x + threadIdx.x;
  if (tid >= nfrag * 32) return;
  int frag = tid >> 5, lane = tid & 31;
  int kt = frag >> 3, t = frag & 7;
  int n  = t * 16 + (lane & 15);
  int kb = (lane & 16) ? 8 : 0;
  FragU A;
#pragma unroll
  for (int e = 0; e < 16; ++e) {
    int koff = (e < 8) ? (e + kb) : (e + 8 + kb);
    int k = kt * 32 + koff;
    A.s[e] = (k < Kdim) ? f2bf(W[(size_t)k * 128 + n]) : (unsigned short)0;
  }
  uint4* o = (uint4*)(dst + (size_t)tid * 16);
  o[0] = A.u[0];
  o[1] = A.u[1];
}

// ---------------------------------------------------------------------------
// Node embedding: nodes = node_features(N,6) @ Wn(6,128) + bn  (trivial VALU)
// ---------------------------------------------------------------------------
__global__ void node_embed_kernel(const float* __restrict__ nf,
                                  const float* __restrict__ Wn,
                                  const float* __restrict__ bn,
                                  float* __restrict__ nodes_f,
                                  unsigned short* __restrict__ nodes_b, int N) {
  int i = blockIdx.x;
  int c = threadIdx.x;
  if (i >= N) return;
  float acc = bn[c];
#pragma unroll
  for (int d = 0; d < 6; ++d) acc += nf[i * 6 + d] * Wn[d * 128 + c];
  nodes_f[(size_t)i * 128 + c] = acc;
  nodes_b[(size_t)i * 128 + c] = f2bf(acc);
}

// ---------------------------------------------------------------------------
// Edge embedding: edges(M,128) = edge_features(M,240) @ We + be  (WMMA GEMM)
// Wave handles 32 rows (2 M-tiles); K padded 240 -> 256.
// ---------------------------------------------------------------------------
__device__ __forceinline__ void cvt8(FragU& A, int e0, const float* p) {
  vf4 a = *(const vf4*)p;
  vf4 b = *(const vf4*)(p + 4);
  vbf4 ha = __builtin_convertvector(a, vbf4);
  vbf4 hb = __builtin_convertvector(b, vbf4);
  *(vbf4*)&A.s[e0]     = ha;
  *(vbf4*)&A.s[e0 + 4] = hb;
}
__device__ __forceinline__ v16bf load_afrag_f32pad(const float* row, int k0, int Kdim) {
  FragU A;
  if (k0 + 8 <= Kdim) cvt8(A, 0, row + k0);
  else {
#pragma unroll
    for (int e = 0; e < 8; ++e) A.s[e] = 0;
  }
  if (k0 + 24 <= Kdim) cvt8(A, 8, row + k0 + 16);
  else {
#pragma unroll
    for (int e = 0; e < 8; ++e) A.s[8 + e] = 0;
  }
  return A.v;
}

__global__ __launch_bounds__(256) void edge_embed_kernel(
    const float* __restrict__ ef, const unsigned short* __restrict__ pWe,
    const float* __restrict__ be, unsigned short* __restrict__ edges_b, int M) {
  const int lane = threadIdx.x & 31;
  const int wave = threadIdx.x >> 5;
  const int mrow = lane & 15;
  const int kb = (lane & 16) ? 8 : 0;
  const int mb = (lane & 16) ? 8 : 0;
  long base = ((long)blockIdx.x * 8 + wave) * 32;
  long r0 = base + mrow;
  long r1 = base + 16 + mrow;
  long r0c = (r0 < M) ? r0 : 0;
  long r1c = (r1 < M) ? r1 : 0;
  const float* rowa = ef + (size_t)r0c * 240;
  const float* rowb = ef + (size_t)r1c * 240;

  v8f Da[8], Db[8];
#pragma unroll
  for (int t = 0; t < 8; ++t) { Da[t] = zero8(); Db[t] = zero8(); }

#pragma unroll
  for (int kt = 0; kt < 8; ++kt) {
    int k0 = (kt << 5) + kb;
    v16bf A0 = load_afrag_f32pad(rowa, k0, 240);
    v16bf A1 = load_afrag_f32pad(rowb, k0, 240);
#pragma unroll
    for (int t = 0; t < 8; ++t) {
      v16bf B = load_bfrag(pWe, kt * 8 + t, lane);
      Da[t] = wmma_bf16(A0, B, Da[t]);
      Db[t] = wmma_bf16(A1, B, Db[t]);
    }
  }
#pragma unroll
  for (int t = 0; t < 8; ++t) {
    float bias = be[t * 16 + mrow];
#pragma unroll
    for (int v = 0; v < 8; ++v) {
      long ra = base + v + mb;
      long rb = base + 16 + v + mb;
      if (ra < M) edges_b[(size_t)ra * 128 + t * 16 + mrow] = f2bf(Da[t][v] + bias);
      if (rb < M) edges_b[(size_t)rb * 128 + t * 16 + mrow] = f2bf(Db[t][v] + bias);
    }
  }
}

// ---------------------------------------------------------------------------
// Message kernel: per node, messages = MLP2([src|nbr|edge]) (K=384->128->128),
// aggregated over 30 neighbors into agg (bf16). One wave per node (32 rows).
// ---------------------------------------------------------------------------
__global__ __launch_bounds__(256) void msg_kernel(
    const unsigned short* __restrict__ nodes_b,
    const unsigned short* __restrict__ edges_b, const int* __restrict__ knn,
    const unsigned short* __restrict__ pW1, const float* __restrict__ b1,
    const unsigned short* __restrict__ pW2, const float* __restrict__ b2,
    unsigned short* __restrict__ agg, int N) {
  __shared__ unsigned short hbuf[8][4096];  // 8 waves x (32 rows x 128) bf16
  const int lane = threadIdx.x & 31;
  const int wave = threadIdx.x >> 5;
  const int node = blockIdx.x * 8 + wave;
  if (node >= N) return;
  const int mrow = lane & 15;
  const int kb = (lane & 16) ? 8 : 0;
  const int mb = (lane & 16) ? 8 : 0;

  const unsigned short* seg0 = nodes_b + (size_t)node * 128;
  int kr0 = mrow;                 // rows 0..15, all valid
  int kr1 = 16 + mrow;            // rows 16..31, valid < 30
  int kr1c = (kr1 < 30) ? kr1 : 29;
  int j0 = knn[node * 30 + kr0];
  int j1 = knn[node * 30 + kr1c];
  const unsigned short* s1a = nodes_b + (size_t)j0 * 128;
  const unsigned short* s1b = nodes_b + (size_t)j1 * 128;
  const unsigned short* s2a = edges_b + ((size_t)node * 30 + kr0) * 128;
  const unsigned short* s2b = edges_b + ((size_t)node * 30 + kr1c) * 128;

  v8f D1a[8], D1b[8];
#pragma unroll
  for (int t = 0; t < 8; ++t) { D1a[t] = zero8(); D1b[t] = zero8(); }

#pragma unroll
  for (int kt = 0; kt < 12; ++kt) {
    const unsigned short* pa;
    const unsigned short* pb;
    if (kt < 4)       { pa = seg0; pb = seg0; }
    else if (kt < 8)  { pa = s1a;  pb = s1b; }
    else              { pa = s2a;  pb = s2b; }
    int k0 = ((kt & 3) << 5) + kb;
    v16bf A0 = load_afrag(pa, k0);
    v16bf A1 = load_afrag(pb, k0);
#pragma unroll
    for (int t = 0; t < 8; ++t) {
      v16bf B = load_bfrag(pW1, kt * 8 + t, lane);
      D1a[t] = wmma_bf16(A0, B, D1a[t]);
      D1b[t] = wmma_bf16(A1, B, D1b[t]);
    }
  }

  unsigned short* hb = hbuf[wave];
#pragma unroll
  for (int t = 0; t < 8; ++t) {
    float bias = b1[t * 16 + mrow];
#pragma unroll
    for (int v = 0; v < 8; ++v) {
      int m = v + mb;
      float xa = fmaxf(D1a[t][v] + bias, 0.f);
      float xb = fmaxf(D1b[t][v] + bias, 0.f);
      hb[m * 128 + t * 16 + mrow] = f2bf(xa);
      hb[(m + 16) * 128 + t * 16 + mrow] = f2bf(xb);
    }
  }
  asm volatile("s_wait_dscnt 0x0" ::: "memory");

  v8f D2a[8], D2b[8];
#pragma unroll
  for (int t = 0; t < 8; ++t) { D2a[t] = zero8(); D2b[t] = zero8(); }
#pragma unroll
  for (int kt = 0; kt < 4; ++kt) {
    int k0 = (kt << 5) + kb;
    v16bf A0 = load_afrag(hb + mrow * 128, k0);
    v16bf A1 = load_afrag(hb + (16 + mrow) * 128, k0);
#pragma unroll
    for (int t = 0; t < 8; ++t) {
      v16bf B = load_bfrag(pW2, kt * 8 + t, lane);
      D2a[t] = wmma_bf16(A0, B, D2a[t]);
      D2b[t] = wmma_bf16(A1, B, D2b[t]);
    }
  }

  // aggregate over K=30 valid rows (half1 rows 30,31 = lanes>=16, v>=6 masked)
#pragma unroll
  for (int t = 0; t < 8; ++t) {
    float bias2 = b2[t * 16 + mrow];
    float s = 0.f;
#pragma unroll
    for (int v = 0; v < 8; ++v) {
      s += D2a[t][v] + bias2;
      bool invalid = (lane >= 16) && (v >= 6);
      if (!invalid) s += D2b[t][v] + bias2;
    }
    s += __shfl_xor(s, 16, 32);
    if (lane < 16) agg[(size_t)node * 128 + t * 16 + lane] = f2bf(s);
  }
}

// ---------------------------------------------------------------------------
// Node update: nodes = LN(nodes + MLP2([nodes|agg])) ; in-place, wave = 32 rows
// ---------------------------------------------------------------------------
__device__ __forceinline__ void node_finish(v8f (&D2)[8], const float* b2,
                                            const float* gamma, const float* beta,
                                            float* nodes_f, unsigned short* nodes_b,
                                            int rowbase, int lane, int N) {
  const int mrow = lane & 15;
  const int mb = (lane & 16) ? 8 : 0;
  float X[8][8], sum[8], sq[8];
#pragma unroll
  for (int v = 0; v < 8; ++v) { sum[v] = 0.f; sq[v] = 0.f; }
#pragma unroll
  for (int t = 0; t < 8; ++t) {
    float bb = b2[t * 16 + mrow];
#pragma unroll
    for (int v = 0; v < 8; ++v) {
      int r = rowbase + v + mb;
      int rc = (r < N) ? r : (N - 1);
      float res = nodes_f[(size_t)rc * 128 + t * 16 + mrow];
      float x = D2[t][v] + bb + res;
      X[t][v] = x;
      sum[v] += x;
      sq[v] += x * x;
    }
  }
#pragma unroll
  for (int v = 0; v < 8; ++v) {
#pragma unroll
    for (int m = 1; m < 16; m <<= 1) {
      sum[v] += __shfl_xor(sum[v], m, 32);
      sq[v]  += __shfl_xor(sq[v],  m, 32);
    }
  }
#pragma unroll
  for (int t = 0; t < 8; ++t) {
    float g = gamma[t * 16 + mrow], bt = beta[t * 16 + mrow];
#pragma unroll
    for (int v = 0; v < 8; ++v) {
      float mean = sum[v] * (1.f / 128.f);
      float var = sq[v] * (1.f / 128.f) - mean * mean;
      float y = (X[t][v] - mean) * rsqrtf(var + 1e-5f) * g + bt;
      int r = rowbase + v + mb;
      if (r < N) {
        nodes_f[(size_t)r * 128 + t * 16 + mrow] = y;
        nodes_b[(size_t)r * 128 + t * 16 + mrow] = f2bf(y);
      }
    }
  }
}

__global__ __launch_bounds__(256) void node_kernel(
    unsigned short* __restrict__ nodes_b, const unsigned short* __restrict__ aggb,
    const unsigned short* __restrict__ pW1, const float* __restrict__ b1,
    const unsigned short* __restrict__ pW2, const float* __restrict__ b2,
    const float* __restrict__ gamma, const float* __restrict__ beta,
    float* __restrict__ nodes_f, int N) {
  __shared__ unsigned short hbuf[8][4096];
  const int lane = threadIdx.x & 31;
  const int wave = threadIdx.x >> 5;
  const int mrow = lane & 15;
  const int kb = (lane & 16) ? 8 : 0;
  const int mb = (lane & 16) ? 8 : 0;
  const int base = (blockIdx.x * 8 + wave) * 32;
  if (base >= N) return;
  int r0 = base + mrow;
  int r1 = base + 16 + mrow;
  int r0c = (r0 < N) ? r0 : (N - 1);
  int r1c = (r1 < N) ? r1 : (N - 1);
  const unsigned short* n0 = nodes_b + (size_t)r0c * 128;
  const unsigned short* n1 = nodes_b + (size_t)r1c * 128;
  const unsigned short* a0 = aggb + (size_t)r0c * 128;
  const unsigned short* a1 = aggb + (size_t)r1c * 128;

  v8f D1a[8], D1b[8];
#pragma unroll
  for (int t = 0; t < 8; ++t) { D1a[t] = zero8(); D1b[t] = zero8(); }
#pragma unroll
  for (int kt = 0; kt < 8; ++kt) {
    const unsigned short* pa = (kt < 4) ? n0 : a0;
    const unsigned short* pb = (kt < 4) ? n1 : a1;
    int k0 = ((kt & 3) << 5) + kb;
    v16bf A0 = load_afrag(pa, k0);
    v16bf A1 = load_afrag(pb, k0);
#pragma unroll
    for (int t = 0; t < 8; ++t) {
      v16bf B = load_bfrag(pW1, kt * 8 + t, lane);
      D1a[t] = wmma_bf16(A0, B, D1a[t]);
      D1b[t] = wmma_bf16(A1, B, D1b[t]);
    }
  }
  unsigned short* hb = hbuf[wave];
#pragma unroll
  for (int t = 0; t < 8; ++t) {
    float bias = b1[t * 16 + mrow];
#pragma unroll
    for (int v = 0; v < 8; ++v) {
      int m = v + mb;
      hb[m * 128 + t * 16 + mrow] = f2bf(fmaxf(D1a[t][v] + bias, 0.f));
      hb[(m + 16) * 128 + t * 16 + mrow] = f2bf(fmaxf(D1b[t][v] + bias, 0.f));
    }
  }
  asm volatile("s_wait_dscnt 0x0" ::: "memory");

  v8f D2a[8], D2b[8];
#pragma unroll
  for (int t = 0; t < 8; ++t) { D2a[t] = zero8(); D2b[t] = zero8(); }
#pragma unroll
  for (int kt = 0; kt < 4; ++kt) {
    int k0 = (kt << 5) + kb;
    v16bf A0 = load_afrag(hb + mrow * 128, k0);
    v16bf A1 = load_afrag(hb + (16 + mrow) * 128, k0);
#pragma unroll
    for (int t = 0; t < 8; ++t) {
      v16bf B = load_bfrag(pW2, kt * 8 + t, lane);
      D2a[t] = wmma_bf16(A0, B, D2a[t]);
      D2b[t] = wmma_bf16(A1, B, D2b[t]);
    }
  }
  node_finish(D2a, b2, gamma, beta, nodes_f, nodes_b, base, lane, N);
  node_finish(D2b, b2, gamma, beta, nodes_f, nodes_b, base + 16, lane, N);
}

// ---------------------------------------------------------------------------
// Edge update: edges = LN(edges + MLP2([src|nbr|edges])) ; in-place bf16
// ---------------------------------------------------------------------------
__device__ __forceinline__ void edge_finish(v8f (&D2)[8], const float* b2,
                                            const float* gamma, const float* beta,
                                            unsigned short* edges_b, size_t rowbase,
                                            size_t rowlast, int lane) {
  const int mrow = lane & 15;
  const int mb = (lane & 16) ? 8 : 0;
  float X[8][8], sum[8], sq[8];
#pragma unroll
  for (int v = 0; v < 8; ++v) { sum[v] = 0.f; sq[v] = 0.f; }
#pragma unroll
  for (int t = 0; t < 8; ++t) {
    float bb = b2[t * 16 + mrow];
#pragma unroll
    for (int v = 0; v < 8; ++v) {
      size_t r = rowbase + v + mb;
      size_t rc = (r <= rowlast) ? r : rowlast;
      float res = bf2f(edges_b[rc * 128 + t * 16 + mrow]);
      float x = D2[t][v] + bb + res;
      X[t][v] = x;
      sum[v] += x;
      sq[v] += x * x;
    }
  }
#pragma unroll
  for (int v = 0; v < 8; ++v) {
#pragma unroll
    for (int m = 1; m < 16; m <<= 1) {
      sum[v] += __shfl_xor(sum[v], m, 32);
      sq[v]  += __shfl_xor(sq[v],  m, 32);
    }
  }
#pragma unroll
  for (int t = 0; t < 8; ++t) {
    float g = gamma[t * 16 + mrow], bt = beta[t * 16 + mrow];
#pragma unroll
    for (int v = 0; v < 8; ++v) {
      float mean = sum[v] * (1.f / 128.f);
      float var = sq[v] * (1.f / 128.f) - mean * mean;
      float y = (X[t][v] - mean) * rsqrtf(var + 1e-5f) * g + bt;
      size_t r = rowbase + v + mb;
      if (r <= rowlast) edges_b[r * 128 + t * 16 + mrow] = f2bf(y);
    }
  }
}

__global__ __launch_bounds__(256) void edge_kernel(
    const unsigned short* __restrict__ nodes_b, unsigned short* __restrict__ edges_b,
    const int* __restrict__ knn,
    const unsigned short* __restrict__ pW1, const float* __restrict__ b1,
    const unsigned short* __restrict__ pW2, const float* __restrict__ b2,
    const float* __restrict__ gamma, const float* __restrict__ beta, int N) {
  __shared__ unsigned short hbuf[8][4096];
  const int lane = threadIdx.x & 31;
  const int wave = threadIdx.x >> 5;
  const int node = blockIdx.x * 8 + wave;
  if (node >= N) return;
  const int mrow = lane & 15;
  const int kb = (lane & 16) ? 8 : 0;
  const int mb = (lane & 16) ? 8 : 0;

  const unsigned short* seg0 = nodes_b + (size_t)node * 128;
  int kr0 = mrow;
  int kr1 = 16 + mrow;
  int kr1c = (kr1 < 30) ? kr1 : 29;
  int j0 = knn[node * 30 + kr0];
  int j1 = knn[node * 30 + kr1c];
  const unsigned short* s1a = nodes_b + (size_t)j0 * 128;
  const unsigned short* s1b = nodes_b + (size_t)j1 * 128;
  const unsigned short* s2a = edges_b + ((size_t)node * 30 + kr0) * 128;
  const unsigned short* s2b = edges_b + ((size_t)node * 30 + kr1c) * 128;

  v8f D1a[8], D1b[8];
#pragma unroll
  for (int t = 0; t < 8; ++t) { D1a[t] = zero8(); D1b[t] = zero8(); }
#pragma unroll
  for (int kt = 0; kt < 12; ++kt) {
    const unsigned short* pa;
    const unsigned short* pb;
    if (kt < 4)      { pa = seg0; pb = seg0; }
    else if (kt < 8) { pa = s1a;  pb = s1b; }
    else             { pa = s2a;  pb = s2b; }
    int k0 = ((kt & 3) << 5) + kb;
    v16bf A0 = load_afrag(pa, k0);
    v16bf A1 = load_afrag(pb, k0);
#pragma unroll
    for (int t = 0; t < 8; ++t) {
      v16bf B = load_bfrag(pW1, kt * 8 + t, lane);
      D1a[t] = wmma_bf16(A0, B, D1a[t]);
      D1b[t] = wmma_bf16(A1, B, D1b[t]);
    }
  }
  unsigned short* hb = hbuf[wave];
#pragma unroll
  for (int t = 0; t < 8; ++t) {
    float bias = b1[t * 16 + mrow];
#pragma unroll
    for (int v = 0; v < 8; ++v) {
      int m = v + mb;
      hb[m * 128 + t * 16 + mrow] = f2bf(fmaxf(D1a[t][v] + bias, 0.f));
      hb[(m + 16) * 128 + t * 16 + mrow] = f2bf(fmaxf(D1b[t][v] + bias, 0.f));
    }
  }
  asm volatile("s_wait_dscnt 0x0" ::: "memory");

  v8f D2a[8], D2b[8];
#pragma unroll
  for (int t = 0; t < 8; ++t) { D2a[t] = zero8(); D2b[t] = zero8(); }
#pragma unroll
  for (int kt = 0; kt < 4; ++kt) {
    int k0 = (kt << 5) + kb;
    v16bf A0 = load_afrag(hb + mrow * 128, k0);
    v16bf A1 = load_afrag(hb + (16 + mrow) * 128, k0);
#pragma unroll
    for (int t = 0; t < 8; ++t) {
      v16bf B = load_bfrag(pW2, kt * 8 + t, lane);
      D2a[t] = wmma_bf16(A0, B, D2a[t]);
      D2b[t] = wmma_bf16(A1, B, D2b[t]);
    }
  }
  size_t rowlast = (size_t)node * 30 + 29;
  edge_finish(D2a, b2, gamma, beta, edges_b, (size_t)node * 30, rowlast, lane);
  edge_finish(D2b, b2, gamma, beta, edges_b, (size_t)node * 30 + 16, rowlast, lane);
}

// ---------------------------------------------------------------------------
// Output head: logits = nodes(N,128) @ Wout(128,20) + bout  (tiny)
// ---------------------------------------------------------------------------
__global__ void out_kernel(const float* __restrict__ nodes_f,
                           const float* __restrict__ Wout,
                           const float* __restrict__ bout, float* __restrict__ out,
                           int N) {
  int idx = blockIdx.x * blockDim.x + threadIdx.x;
  if (idx >= N * 20) return;
  int i = idx / 20, a = idx % 20;
  float acc = bout[a];
#pragma unroll 4
  for (int h = 0; h < 128; ++h) acc += nodes_f[(size_t)i * 128 + h] * Wout[h * 20 + a];
  out[idx] = acc;
}

// ---------------------------------------------------------------------------
// Host-side orchestration
// ---------------------------------------------------------------------------
extern "C" void kernel_launch(void* const* d_in, const int* in_sizes, int n_in,
                              void* d_out, int out_size, void* d_ws, size_t ws_size,
                              hipStream_t stream) {
  (void)n_in; (void)out_size; (void)ws_size;
  const float* node_features = (const float*)d_in[0];
  const float* edge_features = (const float*)d_in[1];
  const int*   knn           = (const int*)d_in[2];
  const float* Wn  = (const float*)d_in[3];
  const float* bn  = (const float*)d_in[4];
  const float* We  = (const float*)d_in[5];
  const float* be  = (const float*)d_in[6];
  const float* mW1 = (const float*)d_in[7];
  const float* mb1 = (const float*)d_in[8];
  const float* mW2 = (const float*)d_in[9];
  const float* mb2 = (const float*)d_in[10];
  const float* nW1 = (const float*)d_in[11];
  const float* nb1 = (const float*)d_in[12];
  const float* nW2 = (const float*)d_in[13];
  const float* nb2 = (const float*)d_in[14];
  const float* eW1 = (const float*)d_in[15];
  const float* eb1 = (const float*)d_in[16];
  const float* eW2 = (const float*)d_in[17];
  const float* eb2 = (const float*)d_in[18];
  const float* g_node = (const float*)d_in[19];
  const float* b_node = (const float*)d_in[20];
  const float* g_edge = (const float*)d_in[21];
  const float* b_edge = (const float*)d_in[22];
  const float* Wout = (const float*)d_in[23];
  const float* bout = (const float*)d_in[24];
  float* out = (float*)d_out;

  const int N = in_sizes[0] / 6;  // 20000
  const int M = N * 30;           // edge rows

  char* ws = (char*)d_ws;
  size_t off = 0;
  auto take = [&](size_t bytes) {
    char* p = ws + off;
    off = (off + bytes + 255) & ~(size_t)255;
    return p;
  };
  float* nodes_f = (float*)take((size_t)N * 128 * 4);
  unsigned short* nodes_b = (unsigned short*)take((size_t)N * 128 * 2);
  unsigned short* aggb    = (unsigned short*)take((size_t)N * 128 * 2);
  unsigned short* edges_b = (unsigned short*)take((size_t)M * 128 * 2);
  unsigned short* pWe = (unsigned short*)take(64 * 1024);
  unsigned short *pmW1[3], *pmW2[3], *pnW1[3], *pnW2[3], *peW1[3], *peW2[3];
  for (int l = 0; l < 3; ++l) {
    pmW1[l] = (unsigned short*)take(96 * 1024);
    pmW2[l] = (unsigned short*)take(32 * 1024);
    pnW1[l] = (unsigned short*)take(64 * 1024);
    pnW2[l] = (unsigned short*)take(32 * 1024);
    peW1[l] = (unsigned short*)take(96 * 1024);
    peW2[l] = (unsigned short*)take(32 * 1024);
  }

  auto pack = [&](const float* W, int Kdim, int ktpad, unsigned short* dst) {
    int nfrag = ktpad * 8;
    int threads = nfrag * 32;
    pack_w_kernel<<<(threads + 255) / 256, 256, 0, stream>>>(W, Kdim, dst, nfrag);
  };
  pack(We, 240, 8, pWe);
  for (int l = 0; l < 3; ++l) {
    pack(mW1 + (size_t)l * 384 * 128, 384, 12, pmW1[l]);
    pack(mW2 + (size_t)l * 128 * 128, 128, 4,  pmW2[l]);
    pack(nW1 + (size_t)l * 256 * 128, 256, 8,  pnW1[l]);
    pack(nW2 + (size_t)l * 128 * 128, 128, 4,  pnW2[l]);
    pack(eW1 + (size_t)l * 384 * 128, 384, 12, peW1[l]);
    pack(eW2 + (size_t)l * 128 * 128, 128, 4,  peW2[l]);
  }

  node_embed_kernel<<<N, 128, 0, stream>>>(node_features, Wn, bn, nodes_f, nodes_b, N);
  edge_embed_kernel<<<(M + 255) / 256, 256, 0, stream>>>(edge_features, pWe, be,
                                                         edges_b, M);
  for (int l = 0; l < 3; ++l) {
    msg_kernel<<<(N + 7) / 8, 256, 0, stream>>>(nodes_b, edges_b, knn, pmW1[l],
                                                mb1 + l * 128, pmW2[l], mb2 + l * 128,
                                                aggb, N);
    node_kernel<<<(N + 255) / 256, 256, 0, stream>>>(
        nodes_b, aggb, pnW1[l], nb1 + l * 128, pnW2[l], nb2 + l * 128,
        g_node + l * 128, b_node + l * 128, nodes_f, N);
    edge_kernel<<<(N + 7) / 8, 256, 0, stream>>>(
        nodes_b, edges_b, knn, peW1[l], eb1 + l * 128, peW2[l], eb2 + l * 128,
        g_edge + l * 128, b_edge + l * 128, N);
  }
  out_kernel<<<(N * 20 + 255) / 256, 256, 0, stream>>>(nodes_f, Wout, bout, out, N);
}
